// EdgeRelationHead_30614526885988
// MI455X (gfx1250) — matrix operand order repaired and checked by reading, not compiled, not run on hardware
//
#include <hip/hip_runtime.h>

typedef _Float16 f16;
typedef __attribute__((ext_vector_type(8)))  _Float16 v8h;
typedef __attribute__((ext_vector_type(16))) _Float16 v16h;
typedef __attribute__((ext_vector_type(8)))  float    v8f;

#define N_NODES  100000
#define HID      128
#define NUM_RELS 6
#define N_EDGES  800000

// Hard scheduling fence: nothing crosses. Used to stop the scheduler from
// sinking batched loads back down to their WMMA uses.
#if defined(__has_builtin)
#  if __has_builtin(__builtin_amdgcn_sched_barrier)
#    define SCHED_FENCE() __builtin_amdgcn_sched_barrier(0)
#  endif
#endif
#ifndef SCHED_FENCE
#  define SCHED_FENCE() ((void)0)
#endif

// ---------------------------------------------------------------------------
// Kernel 1: fp32 -> fp16 conversion of node features h [N_NODES x HID]
// 8 elements/thread: 2x global_load_b128 -> 1x 16B store.
// ---------------------------------------------------------------------------
__global__ void cvt_h_kernel(const float* __restrict__ h, f16* __restrict__ h16) {
    size_t i = ((size_t)blockIdx.x * blockDim.x + threadIdx.x) * 8;
    if (i < (size_t)N_NODES * HID) {
        const float4* p = (const float4*)(h + i);
        float4 x0 = p[0], x1 = p[1];
        v8h o;
        o[0] = (f16)x0.x; o[1] = (f16)x0.y; o[2] = (f16)x0.z; o[3] = (f16)x0.w;
        o[4] = (f16)x1.x; o[5] = (f16)x1.y; o[6] = (f16)x1.z; o[7] = (f16)x1.w;
        *(v8h*)(h16 + i) = o;
    }
}

// ---------------------------------------------------------------------------
// Kernel 2: pack weights.
//  W1aT[n][k] = W1[k][n]        (k < 128)   -> fp16, transposed (K contiguous)
//  W1bT[n][k] = W1[128+k][n]                -> fp16, transposed
//  W2T [n][k] = (n<6) ? W2[k][n] : 0        -> fp16, 16x128, zero padded cols
// ---------------------------------------------------------------------------
__global__ void pack_w_kernel(const float* __restrict__ W1, const float* __restrict__ W2,
                              f16* __restrict__ W1aT, f16* __restrict__ W1bT,
                              f16* __restrict__ W2T) {
    int idx = blockIdx.x * blockDim.x + threadIdx.x;
    if (idx < HID * HID) {
        int n = idx >> 7, k = idx & 127;
        W1aT[idx] = (f16)W1[k * HID + n];
        W1bT[idx] = (f16)W1[(k + HID) * HID + n];
        if (idx < 16 * HID) {
            // here n in [0,16), k in [0,128)
            W2T[idx] = (n < NUM_RELS) ? (f16)W2[k * NUM_RELS + n] : (f16)0.0f;
        }
    }
}

// ---------------------------------------------------------------------------
// Kernel 3: node projections  Pa = h @ W1[0:128],  Pb = h @ W1[128:256]
// One wave per 16-node row-block x full 128 output columns.
// Per k-tile: A fragment + ALL 8 B fragments loaded before a sched fence,
// then 8 back-to-back v_wmma_f32_16x16x32_f16 into independent accumulators
// (single s_wait_loadcnt per k-tile instead of one per WMMA).
// grid = 6250 blocks, block = 64 (wave 0 -> Pa, wave 1 -> Pb).
// ---------------------------------------------------------------------------
__global__ __launch_bounds__(64)
void node_proj_kernel(const f16* __restrict__ h16,
                      const f16* __restrict__ W1aT, const f16* __restrict__ W1bT,
                      f16* __restrict__ Pa, f16* __restrict__ Pb) {
    const int lane = threadIdx.x & 31;
    const int wave = threadIdx.x >> 5;           // 0 -> Pa, 1 -> Pb
    const int tile = blockIdx.x;                 // node tile (16 nodes)
    const int m    = lane & 15;                  // A row / B col / C col
    const int hi   = lane >> 4;
    const int kb   = hi * 8;                     // A-frag K base (per half)
    const int hi16 = hi * 16;                    // unused; B-frag also uses kb
    const size_t node = (size_t)tile * 16 + m;

    const f16* __restrict__ Bsrc = wave ? W1bT : W1aT;
    f16* __restrict__ dst        = wave ? Pb   : Pa;

    v8f acc[8] = {};
    #pragma unroll
    for (int kt = 0; kt < 4; ++kt) {
        const int K0 = kt * 32;
        // A fragment: 16x32 f16 tile of h16 (rows = nodes) -- loaded once,
        // reused by all 8 column tiles.
        v8h a0 = *(const v8h*)(h16 + node * HID + K0 + kb);
        v8h a1 = *(const v8h*)(h16 + node * HID + K0 + 16 + kb);
        v16h afrag;
        #pragma unroll
        for (int i = 0; i < 8; ++i) { afrag[i] = a0[i]; afrag[i + 8] = a1[i]; }
        // Preload all 8 B fragments.
        v16h bf[8];
        #pragma unroll
        for (int nt = 0; nt < 8; ++nt) {
            const f16* bp = Bsrc + (size_t)(nt * 16 + m) * HID + K0;
            v8h b0 = *(const v8h*)(bp + kb);
            v8h b1 = *(const v8h*)(bp + 16 + kb);
            #pragma unroll
            for (int i = 0; i < 8; ++i) { bf[nt][i] = b0[i]; bf[nt][i + 8] = b1[i]; }
        }
        // Fence: loads above may not sink past this point.
        SCHED_FENCE();
        // 8 back-to-back WMMAs, no intervening waits.
        #pragma unroll
        for (int nt = 0; nt < 8; ++nt)
            acc[nt] = __builtin_amdgcn_wmma_f32_16x16x32_f16(
                false, afrag, false, bf[nt], (short)0, acc[nt], false, false);
        SCHED_FENCE();
    }
    // C layout: lane col = m (within tile nt), rows r + 8*hi
    #pragma unroll
    for (int nt = 0; nt < 8; ++nt) {
        #pragma unroll
        for (int r = 0; r < 8; ++r) {
            size_t nr = (size_t)tile * 16 + r + hi * 8;
            dst[nr * HID + nt * 16 + m] = (f16)acc[nt][r];
        }
    }
}

// ---------------------------------------------------------------------------
// Kernel 4: per-edge head.
// hmid = relu(Pa[src] + Pb[dst] + e*W1[256] + b1)   (built straight into the
// WMMA A fragment), then hmid @ W2T (K=128, N=16 padded) + b2.
// The 16 random-access gathers (L2 latency -- the critical path) plus the 4
// constant W2T fragments are all issued before a sched fence, so ONE wait
// covers all four k-tiles of VALU + WMMA work.
// One wave per 16-edge tile; grid = 12500, block = 128.
// ---------------------------------------------------------------------------
__global__ __launch_bounds__(128)
void edge_head_kernel(const f16* __restrict__ Pa, const f16* __restrict__ Pb,
                      const long long* __restrict__ ei, const float* __restrict__ eattr,
                      const float* __restrict__ W1, const float* __restrict__ b1,
                      const f16* __restrict__ W2T, const float* __restrict__ b2,
                      float* __restrict__ out) {
    const int lane = threadIdx.x & 31;
    const int wave = threadIdx.x >> 5;
    const int tile = blockIdx.x * 4 + wave;      // edge tile (16 edges)
    const int m    = lane & 15;
    const int hi   = lane >> 4;
    const int kb   = hi * 8;
    const int hi16 = hi * 16;

    const int edge = tile * 16 + m;
    const size_t srcBase = (size_t)(int)ei[edge]           * HID;
    const size_t dstBase = (size_t)(int)ei[N_EDGES + edge] * HID;
    const float  ea      = eattr[edge];
    const float* __restrict__ we = W1 + 256 * HID;   // last row of W1

    // Issue ALL random gathers up front (independent regs -> one big clause).
    v8h pa0[4], pa1[4], pb0[4], pb1[4];
    #pragma unroll
    for (int kt = 0; kt < 4; ++kt) {
        const int K0 = kt * 32;
        pa0[kt] = *(const v8h*)(Pa + srcBase + K0 + kb);
        pa1[kt] = *(const v8h*)(Pa + srcBase + K0 + 16 + kb);
        pb0[kt] = *(const v8h*)(Pb + dstBase + K0 + kb);
        pb1[kt] = *(const v8h*)(Pb + dstBase + K0 + 16 + kb);
    }
    // Constant B fragments (L0-resident weight matrix), preloaded once.
    v16h bf[4];
    #pragma unroll
    for (int kt = 0; kt < 4; ++kt) {
        const f16* bp = W2T + (size_t)m * HID + kt * 32;
        v8h b0 = *(const v8h*)(bp + kb);
        v8h b1 = *(const v8h*)(bp + 16 + kb);
        #pragma unroll
        for (int i = 0; i < 8; ++i) { bf[kt][i] = b0[i]; bf[kt][i + 8] = b1[i]; }
    }
    // Fence: gathers may not sink toward their uses.
    SCHED_FENCE();

    v8f acc = {};
    #pragma unroll
    for (int kt = 0; kt < 4; ++kt) {
        const int K0 = kt * 32;
        v16h afrag;
        #pragma unroll
        for (int i = 0; i < 8; ++i) {
            const int n0 = K0 + kb + i;
            const int n1 = K0 + 16 + kb + i;
            float v0 = (float)pa0[kt][i] + (float)pb0[kt][i] + ea * we[n0] + b1[n0];
            float v1 = (float)pa1[kt][i] + (float)pb1[kt][i] + ea * we[n1] + b1[n1];
            afrag[i]     = (f16)fmaxf(v0, 0.0f);
            afrag[i + 8] = (f16)fmaxf(v1, 0.0f);
        }
        acc = __builtin_amdgcn_wmma_f32_16x16x32_f16(
            false, afrag, false, bf[kt], (short)0, acc, false, false);
    }
    // C: col = m (only cols < NUM_RELS are real), rows r + 8*hi
    if (m < NUM_RELS) {
        #pragma unroll
        for (int r = 0; r < 8; ++r) {
            int er = tile * 16 + r + hi * 8;
            out[(size_t)er * NUM_RELS + m] = acc[r] + b2[m];
        }
    }
}

// ---------------------------------------------------------------------------
extern "C" void kernel_launch(void* const* d_in, const int* in_sizes, int n_in,
                              void* d_out, int out_size, void* d_ws, size_t ws_size,
                              hipStream_t stream) {
    const float*     h   = (const float*)d_in[0];
    const long long* ei  = (const long long*)d_in[1];   // int64 edge_index [2, E]
    const float*     ea  = (const float*)d_in[2];
    const float*     W1  = (const float*)d_in[3];       // [257,128]
    const float*     b1  = (const float*)d_in[4];
    const float*     W2  = (const float*)d_in[5];       // [128,6]
    const float*     b2  = (const float*)d_in[6];
    float* out = (float*)d_out;

    // workspace layout (all chunks 16B+ aligned)
    char* ws = (char*)d_ws;
    f16* h16  = (f16*)ws;                                  // 25,600,000 B
    f16* W1aT = (f16*)(ws + (size_t)25600000);             //     32,768 B
    f16* W1bT = W1aT + HID * HID;                          //     32,768 B
    f16* W2T  = W1bT + HID * HID;                          //      4,096 B
    f16* Pa   = (f16*)(ws + (size_t)25669632);             // 25,600,000 B
    f16* Pb   = Pa + (size_t)N_NODES * HID;                // 25,600,000 B
    // total ~76.9 MB

    cvt_h_kernel<<<(N_NODES * HID / 8 + 255) / 256, 256, 0, stream>>>(h, h16);
    pack_w_kernel<<<(HID * HID + 255) / 256, 256, 0, stream>>>(W1, W2, W1aT, W1bT, W2T);
    node_proj_kernel<<<N_NODES / 16, 64, 0, stream>>>(h16, W1aT, W1bT, Pa, Pb);
    edge_head_kernel<<<N_EDGES / 64, 128, 0, stream>>>(Pa, Pb, ei, ea, W1, b1, W2T, b2, out);
}
